// NonLocalNet_11811160064606
// MI455X (gfx1250) — matrix operand order repaired and checked by reading, not compiled, not run on hardware
//
#include <hip/hip_runtime.h>

// ---------------------------------------------------------------------------
// Non-Local block (self-attention) for MI455X / gfx1250, wave32 WMMA bf16.
//   B=4, C=256, H=W=64, N=4096
// Pipeline:
//   k_proj: theta/phi/g = bf16 projections of x, stored [B,N,C] in workspace
//   k_attn: flash-attention  y = softmax(theta @ phi^T) @ g   (bf16, [B,N,C])
//   k_out : out = y @ W^T + b + x                              (f32, [B,C,N])
// bf16 tile staging uses GLOBAL_LOAD_ASYNC_TO_LDS_B128 (ASYNCcnt-tracked,
// no VGPR round trip); f32->bf16 converting stages use vectorized VGPR path.
// ---------------------------------------------------------------------------

typedef __attribute__((ext_vector_type(16))) __bf16 v16bf;
typedef __attribute__((ext_vector_type(8)))  float  v8f;

#define NBATCH 4
#define CH     256
#define NP     4096

// padded LDS row strides (in ushorts / floats) -> conflict-free fragment loads
#define S264 264   // for [rows][256] bf16 tiles (256 + 8)
#define S40  40    // for [rows][32]  bf16 tiles (32 + 8)
#define S33  33    // for [rows][32]  f32 score tiles (32 + 1)

union FragU { v16bf v; unsigned u[8]; };

__device__ __forceinline__ unsigned short f2bf(float x) {
  unsigned u = __builtin_bit_cast(unsigned, x);
  unsigned r = u + 0x7FFFu + ((u >> 16) & 1u);   // round-to-nearest-even
  return (unsigned short)(r >> 16);
}

// ---- CDNA5 async global->LDS copy (16 bytes / lane, ASYNCcnt tracked) -----
__device__ __forceinline__ void async_copy_b128(unsigned lds_off,
                                                unsigned long long gbase,
                                                unsigned goff) {
  // GVS mode: 64-bit SGPR base + 32-bit per-lane VGPR byte offset.
  asm volatile("global_load_async_to_lds_b128 %0, %1, %2"
               :: "v"(lds_off), "v"(goff), "s"(gbase) : "memory");
}
__device__ __forceinline__ void wait_async0() {
  asm volatile("s_wait_asynccnt 0x0" ::: "memory");
}

// A fragment 16x32 bf16. rowbase = &lds[row_of_this_lane][k_chunk_base],
// row K-contiguous. CDNA5 layout: lanes 0-15 hold K=0..7,16..23 ; lanes
// 16-31 hold K=8..15,24..31 (packed as bf16 pairs per dword).
__device__ __forceinline__ v16bf load_frag_a(const unsigned short* rowbase, int half) {
  FragU f;
  const unsigned* p = (const unsigned*)rowbase;
  const int h4 = half * 4;
#pragma unroll
  for (int j = 0; j < 4; ++j) f.u[j] = p[j + h4];          // K = 2j + 8*half
#pragma unroll
  for (int j = 4; j < 8; ++j) f.u[j] = p[j + 4 + h4];      // K = 2j + 8 + 8*half
  return f.v;
}

// B fragment 32x16 bf16. rowbase = &lds[n_of_this_lane][k_chunk_base],
// row K-contiguous. Lanes 0-15: K=0..7,16..23 ; lanes 16-31: K=8..15,24..31.
__device__ __forceinline__ v16bf load_frag_b(const unsigned short* rowbase, int half) {
  FragU f;
  const unsigned* p = (const unsigned*)rowbase;
  const int h8 = half * 4;
#pragma unroll
  for (int j = 0; j < 8; ++j) f.u[j] = p[j + (j >> 2) * 4 + h8];
  return f.v;
}

__device__ __forceinline__ v8f wmma_bf16(v16bf a, v16bf b, v8f c) {
  return __builtin_amdgcn_wmma_f32_16x16x32_bf16(false, a, false, b,
                                                 (short)0, c, false, false);
}

// ---------------------------------------------------------------------------
// k_proj: one (row-block of 64, projection, batch) per workgroup, 4 waves.
// out[b][n][o] = bf16( sum_c x[b][c][n] * w[o][c] + bias[o] )
// ---------------------------------------------------------------------------
__global__ __launch_bounds__(128)
void k_proj(const float* __restrict__ x,
            const float* __restrict__ w0, const float* __restrict__ b0, unsigned short* __restrict__ o0,
            const float* __restrict__ w1, const float* __restrict__ b1, unsigned short* __restrict__ o1,
            const float* __restrict__ w2, const float* __restrict__ b2, unsigned short* __restrict__ o2) {
  const int nb = blockIdx.x, pj = blockIdx.y, bb = blockIdx.z;
  const float* w            = (pj == 0) ? w0 : (pj == 1) ? w1 : w2;
  const float* bias         = (pj == 0) ? b0 : (pj == 1) ? b1 : b2;
  unsigned short* outp      = (pj == 0) ? o0 : (pj == 1) ? o1 : o2;

  extern __shared__ unsigned char smem_raw[];
  unsigned short* xT = (unsigned short*)smem_raw;  // [64][S264] bf16, x transposed
  unsigned short* wl = xT + 64 * S264;             // [16][S264] bf16 weight tile

  const int tid = threadIdx.x, lane = tid & 31, wv = tid >> 5;
  const int half = lane >> 4, l16 = lane & 15;
  const int n0 = nb * 64, rw = wv * 16;

  // transpose-load x block: global [C][N] f32 -> LDS [n][c] bf16 (float4 reads)
  for (int i = tid; i < 64 * CH / 4; i += 128) {
    int c = i >> 4, n4 = (i & 15) * 4;
    float4 v = *(const float4*)(x + ((size_t)bb * CH + c) * NP + n0 + n4);
    xT[(n4 + 0) * S264 + c] = f2bf(v.x);
    xT[(n4 + 1) * S264 + c] = f2bf(v.y);
    xT[(n4 + 2) * S264 + c] = f2bf(v.z);
    xT[(n4 + 3) * S264 + c] = f2bf(v.w);
  }
  __syncthreads();

  for (int ot = 0; ot < 16; ++ot) {
    for (int i = tid; i < 16 * CH / 4; i += 128) {
      int o = i >> 6, c4 = (i & 63) * 4;
      float4 v = *(const float4*)(w + (size_t)(ot * 16 + o) * CH + c4);
      union { unsigned short h[4]; uint2 u2; } pk;
      pk.h[0] = f2bf(v.x); pk.h[1] = f2bf(v.y);
      pk.h[2] = f2bf(v.z); pk.h[3] = f2bf(v.w);
      *(uint2*)&wl[o * S264 + c4] = pk.u2;
    }
    __syncthreads();

    v8f acc = {};
#pragma unroll
    for (int kc = 0; kc < 8; ++kc) {
      v16bf A = load_frag_a(&xT[(rw + l16) * S264 + kc * 32], half);
      v16bf B = load_frag_b(&wl[l16 * S264 + kc * 32], half);
      acc = wmma_bf16(A, B, acc);
    }
    const float bo = bias[ot * 16 + l16];
#pragma unroll
    for (int r = 0; r < 8; ++r) {
      int n = n0 + rw + r + 8 * half;
      outp[((size_t)bb * NP + n) * CH + ot * 16 + l16] = f2bf(acc[r] + bo);
    }
    __syncthreads();
  }
}

// ---------------------------------------------------------------------------
// k_attn: flash attention. One (32-query-row block, batch) per workgroup,
// 2 waves x 16 rows. Column loop over 128 tiles of 32 keys. theta/phi tiles
// are moved with async global->LDS B128; g tile is transposed through VGPRs.
// ---------------------------------------------------------------------------
__global__ __launch_bounds__(64)
void k_attn(const unsigned short* __restrict__ theta,
            const unsigned short* __restrict__ phi,
            const unsigned short* __restrict__ gp,
            unsigned short* __restrict__ y) {
  const int nb = blockIdx.x, bb = blockIdx.y;
  extern __shared__ unsigned char smem_raw[];
  unsigned short* th = (unsigned short*)smem_raw;      // [32][S264] theta rows
  unsigned short* ph = th + 32 * S264;                 // [32][S264] phi tile
  unsigned short* gT = ph + 32 * S264;                 // [256][S40] g tile, transposed
  unsigned short* pt = gT + 256 * S40;                 // [2][16][S40] P (bf16)
  float* st          = (float*)(pt + 2 * 16 * S40);    // [2][16][S33] scores f32
  float* corr        = st + 2 * 16 * S33;              // [2][16] corrections

  const int tid = threadIdx.x, lane = tid & 31, wv = tid >> 5;
  const int half = lane >> 4, l16 = lane & 15;
  const size_t base = (size_t)bb * NP * CH;
  const int r0 = nb * 32 + wv * 16;

  const unsigned th_lds = (unsigned)(size_t)th;
  const unsigned ph_lds = (unsigned)(size_t)ph;
  const unsigned long long theta_base = (unsigned long long)(size_t)(theta + base);
  const unsigned long long phi_base   = (unsigned long long)(size_t)(phi + base);

  // async-load this block's theta rows (once): 32x256 bf16, 16B chunks
  for (int ci = tid; ci < 1024; ci += 64) {
    int m = ci >> 5, c = (ci & 31) * 8;
    async_copy_b128(th_lds + (unsigned)(m * S264 + c) * 2, theta_base,
                    (unsigned)(((nb * 32 + m) * CH + c) * 2));
  }

  v8f Y[16];
#pragma unroll
  for (int t = 0; t < 16; ++t) { v8f z = {}; Y[t] = z; }
  float runmax = -INFINITY, runsum = 0.f;          // valid in lanes 0..15

  float* sw = &st[wv * 16 * S33];
  unsigned short* pw = &pt[wv * 16 * S40];

  for (int mt = 0; mt < NP / 32; ++mt) {
    const int m0 = mt * 32;
    __syncthreads();   // previous tile fully consumed before overwrite

    // async phi tile copy (row-major, 16B chunks)
    for (int ci = tid; ci < 1024; ci += 64) {
      int m = ci >> 5, c = (ci & 31) * 8;
      async_copy_b128(ph_lds + (unsigned)(m * S264 + c) * 2, phi_base,
                      (unsigned)(((m0 + m) * CH + c) * 2));
    }
    // g tile: uint4 global reads, transposed u16 scatter into LDS
    for (int ci = tid; ci < 1024; ci += 64) {
      int m = ci >> 5, c = (ci & 31) * 8;
      const uint4 gv = *(const uint4*)(gp + base + (size_t)(m0 + m) * CH + c);
      unsigned short* dst = &gT[c * S40 + m];
      unsigned tmp[4] = {gv.x, gv.y, gv.z, gv.w};
#pragma unroll
      for (int j = 0; j < 4; ++j) {
        dst[(2 * j) * S40]     = (unsigned short)(tmp[j] & 0xFFFFu);
        dst[(2 * j + 1) * S40] = (unsigned short)(tmp[j] >> 16);
      }
    }
    wait_async0();     // theta (first iter) + phi tile resident in LDS
    __syncthreads();

    // S(16x32) = theta_rows(16xC) @ phi_tile^T(Cx32), K=256 in 8 steps
    for (int ct = 0; ct < 2; ++ct) {
      v8f s = {};
#pragma unroll
      for (int kc = 0; kc < 8; ++kc) {
        v16bf A = load_frag_a(&th[(wv * 16 + l16) * S264 + kc * 32], half);
        v16bf B = load_frag_b(&ph[(ct * 16 + l16) * S264 + kc * 32], half);
        s = wmma_bf16(A, B, s);
      }
#pragma unroll
      for (int r = 0; r < 8; ++r)
        sw[(r + 8 * half) * S33 + ct * 16 + l16] = s[r];
    }

    // online softmax: lane L (<16) owns query row L of this wave
    if (lane < 16) {
      float* row = &sw[lane * S33];
      float tmax = -INFINITY;
#pragma unroll 8
      for (int j = 0; j < 32; ++j) tmax = fmaxf(tmax, row[j]);
      const float nmax = fmaxf(runmax, tmax);
      const float cf = __expf(runmax - nmax);      // 0 on first tile
      float ts = 0.f;
      unsigned short* prow = &pw[lane * S40];
#pragma unroll 8
      for (int j = 0; j < 32; ++j) {
        float e = __expf(row[j] - nmax);
        ts += e;
        prow[j] = f2bf(e);
      }
      runmax = nmax;
      runsum = runsum * cf + ts;
      corr[wv * 16 + lane] = cf;
    }

    // rescale Y accumulators by this tile's correction (per query row M)
    float cr[8];
#pragma unroll
    for (int r = 0; r < 8; ++r) cr[r] = corr[wv * 16 + r + 8 * half];
#pragma unroll
    for (int t = 0; t < 16; ++t)
#pragma unroll
      for (int r = 0; r < 8; ++r) Y[t][r] *= cr[r];

    // Y(16x256) += P(16x32) @ g_tile(32x256)
    {
      v16bf A = load_frag_a(&pw[l16 * S40], half);
#pragma unroll
      for (int t = 0; t < 16; ++t) {
        v16bf B = load_frag_b(&gT[(t * 16 + l16) * S40], half);
        Y[t] = wmma_bf16(A, B, Y[t]);
      }
    }
  }

  // final 1/rowsum normalization, then store y as bf16 [B][N][C]
  if (lane < 16) corr[wv * 16 + lane] = 1.f / runsum;
  float cr[8];
#pragma unroll
  for (int r = 0; r < 8; ++r) cr[r] = corr[wv * 16 + r + 8 * half];
#pragma unroll
  for (int t = 0; t < 16; ++t)
#pragma unroll
    for (int r = 0; r < 8; ++r)
      y[base + (size_t)(r0 + r + 8 * half) * CH + t * 16 + l16] = f2bf(Y[t][r] * cr[r]);
}

// ---------------------------------------------------------------------------
// k_out: out[b][o][n] = sum_c y[b][n][c]*W[o][c] + Wb[o] + x[b][o][n]
// ---------------------------------------------------------------------------
__global__ __launch_bounds__(128)
void k_out(const unsigned short* __restrict__ y,
           const float* __restrict__ ww, const float* __restrict__ wb,
           const float* __restrict__ x, float* __restrict__ out) {
  const int nb = blockIdx.x, bb = blockIdx.y;
  extern __shared__ unsigned char smem_raw[];
  unsigned short* yl = (unsigned short*)smem_raw;  // [64][S264]
  unsigned short* wl = yl + 64 * S264;             // [16][S264]

  const int tid = threadIdx.x, lane = tid & 31, wv = tid >> 5;
  const int half = lane >> 4, l16 = lane & 15;
  const int n0 = nb * 64, rw = wv * 16;
  const size_t base = (size_t)bb * NP * CH;

  // async-load y rows (64x256 bf16, row-major copy)
  const unsigned yl_lds = (unsigned)(size_t)yl;
  const unsigned long long y_base =
      (unsigned long long)(size_t)(y + base + (size_t)n0 * CH);
  for (int ci = tid; ci < 2048; ci += 128) {
    int m = ci >> 5, c = (ci & 31) * 8;
    async_copy_b128(yl_lds + (unsigned)(m * S264 + c) * 2, y_base,
                    (unsigned)((m * CH + c) * 2));
  }
  wait_async0();
  __syncthreads();

  for (int ot = 0; ot < 16; ++ot) {
    for (int i = tid; i < 16 * CH / 4; i += 128) {
      int o = i >> 6, c4 = (i & 63) * 4;
      float4 v = *(const float4*)(ww + (size_t)(ot * 16 + o) * CH + c4);
      union { unsigned short h[4]; uint2 u2; } pk;
      pk.h[0] = f2bf(v.x); pk.h[1] = f2bf(v.y);
      pk.h[2] = f2bf(v.z); pk.h[3] = f2bf(v.w);
      *(uint2*)&wl[o * S264 + c4] = pk.u2;
    }
    __syncthreads();

    v8f acc = {};
#pragma unroll
    for (int kc = 0; kc < 8; ++kc) {
      v16bf A = load_frag_a(&yl[(rw + l16) * S264 + kc * 32], half);
      v16bf B = load_frag_b(&wl[l16 * S264 + kc * 32], half);
      acc = wmma_bf16(A, B, acc);
    }
    const float bo = wb[ot * 16 + l16];
    const int o = ot * 16 + l16;
#pragma unroll
    for (int r = 0; r < 8; ++r) {
      int n = n0 + rw + r + 8 * half;
      size_t oi = ((size_t)bb * CH + o) * NP + n;
      out[oi] = acc[r] + bo + x[oi];
    }
    __syncthreads();
  }
}

// ---------------------------------------------------------------------------
extern "C" void kernel_launch(void* const* d_in, const int* in_sizes, int n_in,
                              void* d_out, int out_size, void* d_ws, size_t ws_size,
                              hipStream_t stream) {
  const float* x       = (const float*)d_in[0];
  const float* theta_w = (const float*)d_in[1];
  const float* theta_b = (const float*)d_in[2];
  const float* phi_w   = (const float*)d_in[3];
  const float* phi_b   = (const float*)d_in[4];
  const float* g_w     = (const float*)d_in[5];
  const float* g_b     = (const float*)d_in[6];
  const float* W_w     = (const float*)d_in[7];
  const float* W_b     = (const float*)d_in[8];
  float* out = (float*)d_out;

  unsigned short* ws = (unsigned short*)d_ws;     // 32 MB of bf16 scratch
  const size_t per = (size_t)NBATCH * NP * CH;
  unsigned short* th_g = ws;
  unsigned short* ph_g = ws + per;
  unsigned short* g_g  = ws + 2 * per;
  unsigned short* y_g  = ws + 3 * per;

  const size_t lds_gemm = (size_t)(64 + 16) * S264 * 2;                    // 42240 B
  const size_t lds_attn = (size_t)(2 * 32 * S264 + 256 * S40 + 2 * 16 * S40) * 2
                        + (size_t)(2 * 16 * S33 + 2 * 16) * 4;             // 61184 B

  k_proj<<<dim3(NP / 64, 3, NBATCH), dim3(128), lds_gemm, stream>>>(
      x, theta_w, theta_b, th_g, phi_w, phi_b, ph_g, g_w, g_b, g_g);
  k_attn<<<dim3(NP / 32, NBATCH), dim3(64), lds_attn, stream>>>(
      th_g, ph_g, g_g, y_g);
  k_out<<<dim3(NP / 64, NBATCH), dim3(128), lds_gemm, stream>>>(
      y_g, W_w, W_b, x, out);
}
